// CorrBlock_43404939494137
// MI455X (gfx1250) — matrix operand order repaired and checked by reading, not compiled, not run on hardware
//
#include <hip/hip_runtime.h>
#include <hip/hip_bf16.h>

typedef __attribute__((ext_vector_type(16))) _Float16 v16h;
typedef __attribute__((ext_vector_type(8)))  _Float16 v8h;
typedef __attribute__((ext_vector_type(8)))  float    v8f;

#define BATCH 2
#define CHN   256      // C
#define HW    64       // H == W == feature map side
#define PTOT  4096     // H*W
#define NQ    8        // N query maps
#define CC    324      // 4 levels * 81 taps

// ---------------------------------------------------------------------------
// Stage 1: [B][C][P] f32  ->  [B][P][C] f16   (tiled LDS transpose + convert)
// grid (P/32, C/32, B), block (32, 8)
// ---------------------------------------------------------------------------
__global__ void transpose_convert_kernel(const float* __restrict__ in,
                                         _Float16* __restrict__ out) {
    __shared__ _Float16 tile[32][33];
    const int b  = blockIdx.z;
    const int p0 = blockIdx.x * 32;
    const int c0 = blockIdx.y * 32;
    const float* src = in + (size_t)b * CHN * PTOT;
    _Float16*    dst = out + (size_t)b * PTOT * CHN;

    #pragma unroll
    for (int i = threadIdx.y; i < 32; i += 8) {
        tile[i][threadIdx.x] =
            (_Float16)src[(size_t)(c0 + i) * PTOT + p0 + threadIdx.x];
    }
    __syncthreads();
    #pragma unroll
    for (int i = threadIdx.y; i < 32; i += 8) {
        dst[(size_t)(p0 + i) * CHN + c0 + threadIdx.x] = tile[threadIdx.x][i];
    }
}

// ---------------------------------------------------------------------------
// WMMA fragment loaders (layouts per CDNA5 ISA 7.12.2, wave32)
// A 16x32 f16: lanes 0-15 hold K0-7 | K16-23, lanes 16-31 hold K8-15 | K24-31
// B 32x16 f16: lanes 0-15 hold K0-7 | K16-23, lanes 16-31 hold K8-15 | K24-31 (N = lane&15)
// ---------------------------------------------------------------------------
__device__ __forceinline__ v16h load_a_frag(const _Float16* __restrict__ aRow,
                                            int k0, int grp) {
    const v8h lo = *(const v8h*)(aRow + k0 + grp * 8);
    const v8h hi = *(const v8h*)(aRow + k0 + 16 + grp * 8);
    v16h a;
    #pragma unroll
    for (int j = 0; j < 8; ++j) { a[j] = lo[j]; a[8 + j] = hi[j]; }
    return a;
}

__device__ __forceinline__ v16h load_b_frag(const _Float16* __restrict__ bRow,
                                            int k0, int grp) {
    const v8h lo = *(const v8h*)(bRow + k0 + grp * 8);
    const v8h hi = *(const v8h*)(bRow + k0 + 16 + grp * 8);
    v16h r;
    #pragma unroll
    for (int j = 0; j < 8; ++j) { r[j] = lo[j]; r[8 + j] = hi[j]; }
    return r;
}

// ---------------------------------------------------------------------------
// Stage 2: corr[b][p][q] = (1/16) * sum_c A16[b][p][c] * B16[b][q][c]
// f16 WMMA 16x16x32, f32 accumulate; f16 level-0 store.
// grid = B * (P/16) * (P/512) blocks, 256 threads (8 waves).
// wave w computes the 16 x 64 strip at qTile = qBlk*512 + w*64.
// K-loop is software-pipelined: fragments for k+32 are fetched into fresh
// registers before the WMMAs for k issue, hiding L2 load latency.
// ---------------------------------------------------------------------------
__global__ void corr_gemm_kernel(const _Float16* __restrict__ A16,
                                 const _Float16* __restrict__ B16,
                                 _Float16* __restrict__ L0) {
    const int blk      = blockIdx.x;
    const int qBlkIdx  = blk % (PTOT / 512);
    const int pTileIdx = (blk / (PTOT / 512)) % (PTOT / 16);
    const int b        = blk / ((PTOT / 512) * (PTOT / 16));

    const int wave = threadIdx.x >> 5;
    const int lane = threadIdx.x & 31;
    const int m    = lane & 15;   // M row for A, N col for B/C/D
    const int grp  = lane >> 4;   // lane half-group -> K sub-block

    const int pTile = pTileIdx * 16;
    const int qTile = qBlkIdx * 512 + wave * 64;

    const _Float16* aRow  = A16 + ((size_t)(b * PTOT + pTile + m)) * CHN;
    const _Float16* bRow0 = B16 + ((size_t)(b * PTOT + qTile +  0 + m)) * CHN;
    const _Float16* bRow1 = B16 + ((size_t)(b * PTOT + qTile + 16 + m)) * CHN;
    const _Float16* bRow2 = B16 + ((size_t)(b * PTOT + qTile + 32 + m)) * CHN;
    const _Float16* bRow3 = B16 + ((size_t)(b * PTOT + qTile + 48 + m)) * CHN;

    v8f acc0 = {}, acc1 = {}, acc2 = {}, acc3 = {};

    // Prologue: stage k0 = 0 fragments
    v16h aC  = load_a_frag(aRow, 0, grp);
    v16h b0C = load_b_frag(bRow0, 0, grp);
    v16h b1C = load_b_frag(bRow1, 0, grp);
    v16h b2C = load_b_frag(bRow2, 0, grp);
    v16h b3C = load_b_frag(bRow3, 0, grp);

    #pragma unroll
    for (int k0 = 0; k0 < CHN; k0 += 32) {
        // Prefetch next stage (last iteration re-fetches current; harmless)
        const int kn = (k0 + 32 < CHN) ? (k0 + 32) : k0;
        const v16h aN  = load_a_frag(aRow, kn, grp);
        const v16h b0N = load_b_frag(bRow0, kn, grp);
        const v16h b1N = load_b_frag(bRow1, kn, grp);
        const v16h b2N = load_b_frag(bRow2, kn, grp);
        const v16h b3N = load_b_frag(bRow3, kn, grp);

        acc0 = __builtin_amdgcn_wmma_f32_16x16x32_f16(false, aC, false, b0C, (short)0, acc0, false, false);
        acc1 = __builtin_amdgcn_wmma_f32_16x16x32_f16(false, aC, false, b1C, (short)0, acc1, false, false);
        acc2 = __builtin_amdgcn_wmma_f32_16x16x32_f16(false, aC, false, b2C, (short)0, acc2, false, false);
        acc3 = __builtin_amdgcn_wmma_f32_16x16x32_f16(false, aC, false, b3C, (short)0, acc3, false, false);

        aC = aN; b0C = b0N; b1C = b1N; b2C = b2N; b3C = b3N;
    }

    // Epilogue: D VGPR r -> row M = grp*8 + r, col N = m.  Scale by 1/sqrt(256).
    _Float16* out = L0 + ((size_t)(b * PTOT + pTile)) * (size_t)PTOT + qTile;
    const float scale = 0.0625f;
    #pragma unroll
    for (int r = 0; r < 8; ++r) {
        const size_t rowOff = (size_t)(grp * 8 + r) * PTOT;
        out[rowOff +  0 + m] = (_Float16)(acc0[r] * scale);
        out[rowOff + 16 + m] = (_Float16)(acc1[r] * scale);
        out[rowOff + 32 + m] = (_Float16)(acc2[r] * scale);
        out[rowOff + 48 + m] = (_Float16)(acc3[r] * scale);
    }
}

// ---------------------------------------------------------------------------
// Stage 3: 2x2 stride-2 average pool over the last (h2,w2) dims.
// in: [B*P][2*w2][2*w2] f16 -> out: [B*P][w2][w2] f16 (f32 accumulate)
// ---------------------------------------------------------------------------
__global__ void pool_kernel(const _Float16* __restrict__ in,
                            _Float16* __restrict__ out,
                            int w2, int n) {
    const int i = blockIdx.x * blockDim.x + threadIdx.x;
    if (i >= n) return;
    const int x  = i % w2;
    const int y  = (i / w2) % w2;
    const int bp = i / (w2 * w2);
    const int win = w2 * 2;
    const _Float16* base = in + (size_t)bp * win * win;
    const float v = (float)base[(2 * y) * win + 2 * x]
                  + (float)base[(2 * y) * win + 2 * x + 1]
                  + (float)base[(2 * y + 1) * win + 2 * x]
                  + (float)base[(2 * y + 1) * win + 2 * x + 1];
    out[i] = (_Float16)(v * 0.25f);
}

// ---------------------------------------------------------------------------
// Stage 4: 81-tap bilinear lookup per (b,n,h,w) per level.
// Replicates reference math exactly: x = clamp(cx/2^i + d[ky]), y = clamp(cy/2^i + d[kx]).
// Output written in raw (B,N,H,W,CC) row-major order == final reshape.
// ---------------------------------------------------------------------------
__global__ void lookup_kernel(const float* __restrict__ coords,
                              const _Float16* __restrict__ p0,
                              const _Float16* __restrict__ p1,
                              const _Float16* __restrict__ p2,
                              const _Float16* __restrict__ p3,
                              float* __restrict__ out) {
    const int tid = blockIdx.x * blockDim.x + threadIdx.x;   // B*N*H*W = 65536
    const int wq = tid & 63;
    const int h  = (tid >> 6) & 63;
    const int n  = (tid >> 12) & 7;
    const int b  = tid >> 15;

    const size_t cbase = ((((size_t)(b * NQ + n)) * 2) * HW + h) * HW + wq;
    const float cx = coords[cbase];                   // channel 0 = x
    const float cy = coords[cbase + (size_t)HW * HW]; // channel 1 = y

    const int p = h * HW + wq;
    const _Float16* pyr[4] = {p0, p1, p2, p3};
    float* o = out + (size_t)tid * CC;

    #pragma unroll
    for (int lev = 0; lev < 4; ++lev) {
        const int w2 = HW >> lev;
        const _Float16* row = pyr[lev] + ((size_t)(b * PTOT + p)) * (w2 * w2);
        const float inv = 1.0f / (float)(1 << lev);
        const float bx = cx * inv;
        const float by = cy * inv;
        const float lim = (float)(w2 - 1);
        for (int t = 0; t < 81; ++t) {
            const int ky = t / 9;
            const int kx = t - ky * 9;
            const float x = fminf(fmaxf(bx + (float)(ky - 4), 0.0f), lim);
            const float y = fminf(fmaxf(by + (float)(kx - 4), 0.0f), lim);
            const float x0 = floorf(x), y0 = floorf(y);
            const float fx = x - x0,  fy = y - y0;
            const int x0i = (int)x0, y0i = (int)y0;
            const int x1i = min(x0i + 1, w2 - 1);
            const int y1i = min(y0i + 1, w2 - 1);
            const float c00 = (float)row[y0i * w2 + x0i];
            const float c01 = (float)row[y0i * w2 + x1i];
            const float c10 = (float)row[y1i * w2 + x0i];
            const float c11 = (float)row[y1i * w2 + x1i];
            o[lev * 81 + t] = c00 * (1.0f - fy) * (1.0f - fx)
                            + c01 * (1.0f - fy) * fx
                            + c10 * fy * (1.0f - fx)
                            + c11 * fy * fx;
        }
    }
}

// ---------------------------------------------------------------------------
// Workspace layout (bytes):
//   A16 : 0                        size  4,194,304   (B*P*C f16)
//   B16 : 4,194,304                size  4,194,304
//   L0  : 8,388,608                size 67,108,864   (B*P*4096 f16)
//   L1  : 75,497,472               size 16,777,216
//   L2  : 92,274,688               size  4,194,304
//   L3  : 96,468,992               size  1,048,576
// total ~93 MB  (f16 pyramid fits in the 192 MB L2)
// ---------------------------------------------------------------------------
extern "C" void kernel_launch(void* const* d_in, const int* in_sizes, int n_in,
                              void* d_out, int out_size, void* d_ws, size_t ws_size,
                              hipStream_t stream) {
    const float* fmap1  = (const float*)d_in[0];
    const float* fmap2  = (const float*)d_in[1];
    const float* coords = (const float*)d_in[2];
    float* out = (float*)d_out;

    char* ws = (char*)d_ws;
    _Float16* A16 = (_Float16*)(ws);
    _Float16* B16 = (_Float16*)(ws + 4194304ull);
    _Float16* L0  = (_Float16*)(ws + 8388608ull);
    _Float16* L1  = (_Float16*)(ws + 75497472ull);
    _Float16* L2  = (_Float16*)(ws + 92274688ull);
    _Float16* L3  = (_Float16*)(ws + 96468992ull);

    // Stage 1: transpose + f32->f16 for both feature maps
    {
        dim3 grid(PTOT / 32, CHN / 32, BATCH);
        dim3 block(32, 8);
        transpose_convert_kernel<<<grid, block, 0, stream>>>(fmap1, A16);
        transpose_convert_kernel<<<grid, block, 0, stream>>>(fmap2, B16);
    }

    // Stage 2: WMMA all-pairs correlation (level 0)
    {
        dim3 grid(BATCH * (PTOT / 16) * (PTOT / 512));  // 4096 blocks
        corr_gemm_kernel<<<grid, 256, 0, stream>>>(A16, B16, L0);
    }

    // Stage 3: pyramid levels 1..3
    {
        int n1 = BATCH * PTOT * 32 * 32;
        pool_kernel<<<(n1 + 255) / 256, 256, 0, stream>>>(L0, L1, 32, n1);
        int n2 = BATCH * PTOT * 16 * 16;
        pool_kernel<<<(n2 + 255) / 256, 256, 0, stream>>>(L1, L2, 16, n2);
        int n3 = BATCH * PTOT * 8 * 8;
        pool_kernel<<<(n3 + 255) / 256, 256, 0, stream>>>(L2, L3, 8, n3);
    }

    // Stage 4: bilinear lookup, fused over all 4 levels
    {
        int nq = BATCH * NQ * HW * HW;  // 65536 threads
        lookup_kernel<<<(nq + 255) / 256, 256, 0, stream>>>(coords, L0, L1, L2, L3, out);
    }
}
